// MaskedMultiHeadSelfAttention_82197084110976
// MI455X (gfx1250) — matrix operand order repaired and checked by reading, not compiled, not run on hardware
//
#include <hip/hip_runtime.h>
#include <math.h>

// MI455X / gfx1250, wave32. Matmuls via v_wmma_f32_16x16x32_f16, f16 operands
// double-buffered through LDS; staging uses GLOBAL_LOAD_ASYNC_TO_LDS_B128
// (ASYNCcnt) when the toolchain exposes it, else reg-staged b128 copies.

typedef __attribute__((ext_vector_type(16))) _Float16 v16h;
typedef __attribute__((ext_vector_type(8)))  _Float16 h8;
typedef __attribute__((ext_vector_type(8)))  float    v8f;

#define D_MODEL 1024
#define NUM_HEADS 16
#define D_K 64
#define BATCH 2
#define SEQ 2048
#define QKV_LD (3 * D_MODEL)
#define MTOT (BATCH * SEQ)

#if __has_builtin(__builtin_amdgcn_global_load_async_to_lds_b128) && \
    __has_builtin(__builtin_amdgcn_s_wait_asynccnt)
#define HAVE_ASYNC_LDS 1
#else
#define HAVE_ASYNC_LDS 0
#endif

#if HAVE_ASYNC_LDS
// Builtin signature (from hipcc diagnostic): (int4 AS1*, int4 AS3*, Ii, Ii)
typedef int i4v __attribute__((vector_size(16)));
typedef __attribute__((address_space(1))) i4v* gptr_i4;
typedef __attribute__((address_space(3))) i4v* lptr_i4;
__device__ __forceinline__ void async_copy_b128(void* lds, const void* gl) {
    __builtin_amdgcn_global_load_async_to_lds_b128(
        (gptr_i4)gl, (lptr_i4)lds, 0, 0);
}
#endif

// ---------------------------------------------------------------------------
// One-time precision / layout conversion kernels (tiny vs. GEMM cost).
// ---------------------------------------------------------------------------
__global__ void cvt_f32_to_f16(const float* __restrict__ in,
                               _Float16* __restrict__ out, int n) {
    int i = blockIdx.x * blockDim.x + threadIdx.x;
    if (i < n) out[i] = (_Float16)in[i];
}

// W [K,N] f32 row-major -> WT [N,K] f16 (GEMM B-operand contiguous in K)
__global__ void transpose_w_f16(const float* __restrict__ W,
                                _Float16* __restrict__ WT, int K, int N) {
    int i = blockIdx.x * blockDim.x + threadIdx.x;
    if (i >= K * N) return;
    int k = i / N, n = i - k * N;
    WT[(size_t)n * K + k] = (_Float16)W[i];
}

// qkvh [B,T,3C] f16 -> Vt [B,H,dk,T] f16 (V LDS staging contiguous in t)
__global__ void transpose_v_f16(const _Float16* __restrict__ qkvh,
                                _Float16* __restrict__ Vt) {
    int i = blockIdx.x * blockDim.x + threadIdx.x;
    if (i >= BATCH * SEQ * D_MODEL) return;
    int c = i % D_MODEL;
    int t = (i / D_MODEL) % SEQ;
    int b = i / (D_MODEL * SEQ);
    _Float16 v = qkvh[((size_t)(b * SEQ + t)) * QKV_LD + 2 * D_MODEL + c];
    Vt[((size_t)(b * NUM_HEADS + c / D_K) * D_K + (c % D_K)) * SEQ + t] = v;
}

// ---------------------------------------------------------------------------
// GEMM: C[M,N] = A[M,K] @ BT[N,K]^T + bias.  A,BT f16; C f32 or f16.
// Block = 256 threads (8 waves), tile 128(M) x 64(N), K-step 32,
// double-buffered LDS: next tile staged while current tile feeds WMMA.
// ---------------------------------------------------------------------------
__global__ __launch_bounds__(256) void gemm_f16_wmma(
    const _Float16* __restrict__ A,   // [M,K]
    const _Float16* __restrict__ BT,  // [N,K]
    const float* __restrict__ bias,   // [N]
    float* __restrict__ Cf,           // [M,N] or nullptr
    _Float16* __restrict__ Ch,        // [M,N] or nullptr
    int M, int N, int K)
{
    __shared__ __align__(16) _Float16 Ash[2][128][40];  // 80B rows, 16B-aligned
    __shared__ __align__(16) _Float16 Bsh[2][64][40];

    const int tid  = threadIdx.x;
    const int lane = tid & 31, wave = tid >> 5;
    const int mrow = lane & 15, hi = lane >> 4;
    const int m_blk = blockIdx.y * 128;
    const int n_blk = blockIdx.x * 64;

    // Per-thread staging coordinates (all b128 granules)
    const int ar0 = tid >> 2,        aq0 = (tid & 3) * 8;        // A granule 0
    const int ar1 = 64 + (tid >> 2), aq1 = (tid & 3) * 8;        // A granule 1
    const int br  = tid >> 2,        bq  = (tid & 3) * 8;        // B granule

    const _Float16* Ap0 = A  + (size_t)(m_blk + ar0) * K + aq0;
    const _Float16* Ap1 = A  + (size_t)(m_blk + ar1) * K + aq1;
    const _Float16* Bp  = BT + (size_t)(n_blk + br)  * K + bq;

    const v8f vzero = {0.f, 0.f, 0.f, 0.f, 0.f, 0.f, 0.f, 0.f};
    v8f acc[4];
    #pragma unroll
    for (int t = 0; t < 4; ++t) acc[t] = vzero;

    const int nsteps = K >> 5;

    // Prologue: stage tile 0 into buffer 0 (synchronous reg staging).
    {
        h8 ra0 = *(const h8*)Ap0;
        h8 ra1 = *(const h8*)Ap1;
        h8 rb  = *(const h8*)Bp;
        *(h8*)&Ash[0][ar0][aq0] = ra0;
        *(h8*)&Ash[0][ar1][aq1] = ra1;
        *(h8*)&Bsh[0][br][bq]   = rb;
    }
    __syncthreads();

    for (int step = 0; step < nsteps; ++step) {
        const int cur = step & 1, nxt = cur ^ 1;
        const int k1 = (step + 1) << 5;
        const bool havenext = (step + 1) < nsteps;

#if HAVE_ASYNC_LDS
        if (havenext) {   // direct global -> LDS DMA into the other buffer
            async_copy_b128(&Ash[nxt][ar0][aq0], Ap0 + k1);
            async_copy_b128(&Ash[nxt][ar1][aq1], Ap1 + k1);
            async_copy_b128(&Bsh[nxt][br][bq],   Bp  + k1);
        }
#else
        h8 ra0, ra1, rb;
        if (havenext) {   // issue next-tile global loads before compute
            ra0 = *(const h8*)(Ap0 + k1);
            ra1 = *(const h8*)(Ap1 + k1);
            rb  = *(const h8*)(Bp  + k1);
        }
#endif

        // ---- compute from buffer `cur` ----
        v16h a;
        {
            h8 vlo = *(const h8*)&Ash[cur][wave * 16 + mrow][8 * hi];
            h8 vhi = *(const h8*)&Ash[cur][wave * 16 + mrow][16 + 8 * hi];
            #pragma unroll
            for (int j = 0; j < 8; ++j) { a[j] = vlo[j]; a[8 + j] = vhi[j]; }
        }
        #pragma unroll
        for (int t = 0; t < 4; ++t) {
            h8 vlo = *(const h8*)&Bsh[cur][t * 16 + mrow][8 * hi];
            h8 vhi = *(const h8*)&Bsh[cur][t * 16 + mrow][16 + 8 * hi];
            v16h bm;
            #pragma unroll
            for (int j = 0; j < 8; ++j) { bm[j] = vlo[j]; bm[8 + j] = vhi[j]; }
            acc[t] = __builtin_amdgcn_wmma_f32_16x16x32_f16(
                false, a, false, bm, (short)0, acc[t], false, false);
        }

#if HAVE_ASYNC_LDS
        if (havenext) __builtin_amdgcn_s_wait_asynccnt(0);
#else
        if (havenext) {
            *(h8*)&Ash[nxt][ar0][aq0] = ra0;
            *(h8*)&Ash[nxt][ar1][aq1] = ra1;
            *(h8*)&Bsh[nxt][br][bq]   = rb;
        }
#endif
        __syncthreads();
    }

    // ---- epilogue: C/D layout row = r + 8*hi, col = mrow ----
    #pragma unroll
    for (int t = 0; t < 4; ++t) {
        const int n = n_blk + t * 16 + mrow;
        const float bv = bias[n];
        #pragma unroll
        for (int r = 0; r < 8; ++r) {
            const size_t off = (size_t)(m_blk + wave * 16 + r + 8 * hi) * N + n;
            const float val = acc[t][r] + bv;
            if (Ch) Ch[off] = (_Float16)val;
            else    Cf[off] = val;
        }
    }
}

// ---------------------------------------------------------------------------
// Causal flash attention. Block = 256 threads (8 waves) covering 128 query
// rows of one (batch, head); 32-key K/V tiles double-buffered in LDS and
// shared by all 8 waves. Online softmax in fp32, matmuls in f16 WMMA.
// ---------------------------------------------------------------------------
__global__ __launch_bounds__(256) void flash_attn_wmma(
    const _Float16* __restrict__ qkvh,  // [B,T,3C] f16 (Q at 0, K at +1024)
    const _Float16* __restrict__ Vt,    // [B,H,dk,T] f16
    _Float16* __restrict__ attnh)       // [B,T,C] f16
{
    __shared__ __align__(16) _Float16 Ksh[2][32][72];      // [key][d], 144B rows
    __shared__ __align__(16) _Float16 Vsh[2][64][40];      // [d][key], 80B rows
    __shared__ __align__(16) _Float16 Pbuf[8][16][40];     // per-wave P bounce

    const int tid  = threadIdx.x;
    const int lane = tid & 31, wave = tid >> 5;
    const int mrow = lane & 15, hi = lane >> 4;
    const int qbase = blockIdx.x * 128;
    const int h = blockIdx.y, b = blockIdx.z;
    const int qrow0 = qbase + wave * 16;

    // Per-thread staging coordinates
    const int kkey = tid >> 3, kq = (tid & 7) * 8;          // Ksh granule
    const int vd   = tid >> 2, vq = (tid & 3) * 8;          // Vsh granule
    const _Float16* Kp = qkvh + ((size_t)(b * SEQ) + kkey) * QKV_LD
                         + D_MODEL + h * D_K + kq;          // + kb*QKV_LD per tile
    const _Float16* Vp = Vt + ((size_t)(b * NUM_HEADS + h) * D_K + vd) * SEQ + vq;

    // Q tile (16x64) for this wave: two A-layout operands, pure b128 loads.
    v16h qa[2];
    {
        const _Float16* Qrow =
            qkvh + ((size_t)(b * SEQ) + qrow0 + mrow) * QKV_LD + h * D_K;
        #pragma unroll
        for (int s = 0; s < 2; ++s) {
            h8 vlo = *(const h8*)(Qrow + s * 32 + 8 * hi);
            h8 vhi = *(const h8*)(Qrow + s * 32 + 16 + 8 * hi);
            #pragma unroll
            for (int j = 0; j < 8; ++j) { qa[s][j] = vlo[j]; qa[s][8 + j] = vhi[j]; }
        }
    }

    const v8f vzero = {0.f, 0.f, 0.f, 0.f, 0.f, 0.f, 0.f, 0.f};
    v8f oacc[4];
    #pragma unroll
    for (int t = 0; t < 4; ++t) oacc[t] = vzero;
    float mi[8], li[8];
    #pragma unroll
    for (int r = 0; r < 8; ++r) { mi[r] = -INFINITY; li[r] = 0.f; }

    const int qmax  = qrow0 + 15;                  // this wave's causal horizon
    const int nkb   = (qbase + 128) >> 5;          // #key tiles for the block
    const float scale = 0.125f;                    // 1/sqrt(64)

    // Prologue: stage key tile 0 into buffer 0.
    {
        h8 rk = *(const h8*)Kp;
        h8 rv = *(const h8*)Vp;
        *(h8*)&Ksh[0][kkey][kq] = rk;
        *(h8*)&Vsh[0][vd][vq]   = rv;
    }
    __syncthreads();

    for (int step = 0; step < nkb; ++step) {
        const int kb  = step << 5;
        const int cur = step & 1, nxt = cur ^ 1;
        const bool havenext = (step + 1) < nkb;

#if HAVE_ASYNC_LDS
        if (havenext) {
            async_copy_b128(&Ksh[nxt][kkey][kq], Kp + (size_t)(kb + 32) * QKV_LD);
            async_copy_b128(&Vsh[nxt][vd][vq],   Vp + kb + 32);
        }
#else
        h8 rk, rv;
        if (havenext) {
            rk = *(const h8*)(Kp + (size_t)(kb + 32) * QKV_LD);
            rv = *(const h8*)(Vp + kb + 32);
        }
#endif

        if (kb <= qmax) {   // wave-uniform: EXEC stays all-ones inside
            // ---- S = Q @ K^T : two 16x16 score tiles over 32 keys ----
            v8f s0 = vzero, s1 = vzero;
            #pragma unroll
            for (int s = 0; s < 2; ++s) {   // d-slice 0..31 / 32..63
                v16h k0m, k1m;
                {
                    h8 vlo = *(const h8*)&Ksh[cur][mrow][s * 32 + 8 * hi];
                    h8 vhi = *(const h8*)&Ksh[cur][mrow][s * 32 + 16 + 8 * hi];
                    #pragma unroll
                    for (int j = 0; j < 8; ++j) { k0m[j] = vlo[j]; k0m[8 + j] = vhi[j]; }
                }
                {
                    h8 vlo = *(const h8*)&Ksh[cur][16 + mrow][s * 32 + 8 * hi];
                    h8 vhi = *(const h8*)&Ksh[cur][16 + mrow][s * 32 + 16 + 8 * hi];
                    #pragma unroll
                    for (int j = 0; j < 8; ++j) { k1m[j] = vlo[j]; k1m[8 + j] = vhi[j]; }
                }
                s0 = __builtin_amdgcn_wmma_f32_16x16x32_f16(
                    false, qa[s], false, k0m, (short)0, s0, false, false);
                s1 = __builtin_amdgcn_wmma_f32_16x16x32_f16(
                    false, qa[s], false, k1m, (short)0, s1, false, false);
            }

            // ---- causal mask + online softmax (fp32) ----
            const int col0 = kb + mrow;
            const int col1 = col0 + 16;
            #pragma unroll
            for (int r = 0; r < 8; ++r) {
                const int row = qrow0 + r + 8 * hi;
                const float v0 = (col0 <= row) ? s0[r] * scale : -INFINITY;
                const float v1 = (col1 <= row) ? s1[r] * scale : -INFINITY;
                float v = fmaxf(v0, v1);
                #pragma unroll
                for (int off = 1; off < 16; off <<= 1)
                    v = fmaxf(v, __shfl_xor(v, off, 32));
                const float mnew = fmaxf(mi[r], v);
                const float corr = __expf(mi[r] - mnew);
                const float e0 = __expf(v0 - mnew);
                const float e1 = __expf(v1 - mnew);
                float ts = e0 + e1;
                #pragma unroll
                for (int off = 1; off < 16; off <<= 1)
                    ts += __shfl_xor(ts, off, 32);
                li[r] = li[r] * corr + ts;
                mi[r] = mnew;
                #pragma unroll
                for (int t = 0; t < 4; ++t) oacc[t][r] *= corr;
                Pbuf[wave][r + 8 * hi][mrow]      = (_Float16)e0;
                Pbuf[wave][r + 8 * hi][16 + mrow] = (_Float16)e1;
            }

            // intra-wave LDS ordering for the C-layout -> A-layout bounce
            asm volatile("s_wait_dscnt 0x0" ::: "memory");

            v16h pa;
            {
                h8 vlo = *(const h8*)&Pbuf[wave][mrow][8 * hi];
                h8 vhi = *(const h8*)&Pbuf[wave][mrow][16 + 8 * hi];
                #pragma unroll
                for (int j = 0; j < 8; ++j) { pa[j] = vlo[j]; pa[8 + j] = vhi[j]; }
            }

            // ---- O += P @ V ----
            #pragma unroll
            for (int t = 0; t < 4; ++t) {
                h8 vlo = *(const h8*)&Vsh[cur][t * 16 + mrow][8 * hi];
                h8 vhi = *(const h8*)&Vsh[cur][t * 16 + mrow][16 + 8 * hi];
                v16h vm;
                #pragma unroll
                for (int j = 0; j < 8; ++j) { vm[j] = vlo[j]; vm[8 + j] = vhi[j]; }
                oacc[t] = __builtin_amdgcn_wmma_f32_16x16x32_f16(
                    false, pa, false, vm, (short)0, oacc[t], false, false);
            }
        }

#if HAVE_ASYNC_LDS
        if (havenext) __builtin_amdgcn_s_wait_asynccnt(0);
#else
        if (havenext) {
            *(h8*)&Ksh[nxt][kkey][kq] = rk;
            *(h8*)&Vsh[nxt][vd][vq]   = rv;
        }
#endif
        __syncthreads();
    }

    // ---- finalize: O / l -> f16 concat layout ----
    #pragma unroll
    for (int t = 0; t < 4; ++t) {
        const int n = h * D_K + t * 16 + mrow;
        #pragma unroll
        for (int r = 0; r < 8; ++r) {
            const int row = qrow0 + r + 8 * hi;
            attnh[((size_t)(b * SEQ) + row) * D_MODEL + n] =
                (_Float16)(oacc[t][r] / li[r]);
        }
    }
}

// ---------------------------------------------------------------------------
extern "C" void kernel_launch(void* const* d_in, const int* in_sizes, int n_in,
                              void* d_out, int out_size, void* d_ws, size_t ws_size,
                              hipStream_t stream) {
    (void)in_sizes; (void)n_in; (void)out_size; (void)ws_size;

    const float* x    = (const float*)d_in[0];   // [2,2048,1024]
    const float* Wqkv = (const float*)d_in[1];   // [1024,3072]
    const float* bqkv = (const float*)d_in[2];   // [3072]
    const float* Wout = (const float*)d_in[3];   // [1024,1024]
    const float* bout = (const float*)d_in[4];   // [1024]
    float*       out  = (float*)d_out;           // [2,2048,1024] f32

    // Workspace carve-up (f16): ~56 MB total
    _Float16* xh     = (_Float16*)d_ws;                          // 4096*1024
    _Float16* WqkvT  = xh    + (size_t)MTOT * D_MODEL;           // 3072*1024
    _Float16* WoutT  = WqkvT + (size_t)QKV_LD * D_MODEL;         // 1024*1024
    _Float16* qkvh   = WoutT + (size_t)D_MODEL * D_MODEL;        // 4096*3072
    _Float16* Vt     = qkvh  + (size_t)MTOT * QKV_LD;            // 2*16*64*2048
    _Float16* attnh  = Vt    + (size_t)BATCH * NUM_HEADS * D_K * SEQ; // 4096*1024

    const int nX  = MTOT * D_MODEL;
    const int nWq = D_MODEL * QKV_LD;
    const int nWo = D_MODEL * D_MODEL;

    cvt_f32_to_f16<<<(nX + 255) / 256, 256, 0, stream>>>(x, xh, nX);
    transpose_w_f16<<<(nWq + 255) / 256, 256, 0, stream>>>(Wqkv, WqkvT, D_MODEL, QKV_LD);
    transpose_w_f16<<<(nWo + 255) / 256, 256, 0, stream>>>(Wout, WoutT, D_MODEL, D_MODEL);

    // 1) QKV projection -> f16 intermediate
    {
        dim3 grid(QKV_LD / 64, MTOT / 128);      // (48, 32)
        gemm_f16_wmma<<<grid, 256, 0, stream>>>(xh, WqkvT, bqkv,
                                                nullptr, qkvh, MTOT, QKV_LD, D_MODEL);
    }
    // 2) V transpose + causal flash attention -> f16 concat
    transpose_v_f16<<<(nX + 255) / 256, 256, 0, stream>>>(qkvh, Vt);
    {
        dim3 grid(SEQ / 128, NUM_HEADS, BATCH);  // (16, 16, 2)
        flash_attn_wmma<<<grid, 256, 0, stream>>>(qkvh, Vt, attnh);
    }
    // 3) output projection -> f32 result
    {
        dim3 grid(D_MODEL / 64, MTOT / 128);     // (16, 32)
        gemm_f16_wmma<<<grid, 256, 0, stream>>>(attnh, WoutT, bout,
                                                out, nullptr, MTOT, D_MODEL, D_MODEL);
    }
}
